// MultiHeadAttentionCombinedQKV_69879117906639
// MI455X (gfx1250) — matrix operand order repaired and checked by reading, compiled, not run on hardware
//
#include <hip/hip_runtime.h>
#include <cstdint>

// ---------------------------------------------------------------------------
// MultiHeadAttentionCombinedQKV for MI455X (gfx1250), wave32 + WMMA.
// B=2, S=2048, D_IN=D_OUT=1024, H=16, Dh=64. fp32 in/out, f16 WMMA compute
// with f32 accumulation (workload is matrix-op bound: ~52 GFLOP vs ~50MB).
// ---------------------------------------------------------------------------

typedef __attribute__((ext_vector_type(16))) _Float16 v16h;
typedef __attribute__((ext_vector_type(8)))  _Float16 v8h;
typedef __attribute__((ext_vector_type(8)))  float    v8f;

#define S_LEN   2048
#define D_MODEL 1024
#define NHEAD   16
#define HDIM    64
#define BATCH   2
#define BS_ROWS (BATCH * S_LEN)          // 4096 GEMM rows
#define QKV_ELEMS ((size_t)3 * BATCH * NHEAD * S_LEN * HDIM)  // f16 elems

__device__ __forceinline__ v8f wmma16(v16h a, v16h b, v8f c) {
  // v_wmma_f32_16x16x32_f16: D = A(16x32) * B(32x16) + C(16x16 f32)
  return __builtin_amdgcn_wmma_f32_16x16x32_f16(
      /*neg_a=*/false, a, /*neg_b=*/false, b,
      /*c_mod=*/(short)0, c, /*reuse_a=*/false, /*reuse_b=*/false);
}

// A-fragment (16x32 f16): lane holds row m=lane&15; K runs
// contiguous [16*hi .. 16*hi+15] packed 2-per-VGPR.
__device__ __forceinline__ v16h load_a_frag(const _Float16* row, int hi) {
  v8h lo = *(const v8h*)(row + hi * 16);
  v8h hh = *(const v8h*)(row + hi * 16 + 8);
  v16h a;
#pragma unroll
  for (int i = 0; i < 8; ++i) { a[i] = lo[i]; a[i + 8] = hh[i]; }
  return a;
}

__device__ __forceinline__ float rowmax16(float v) {
#pragma unroll
  for (int m = 8; m >= 1; m >>= 1) v = fmaxf(v, __shfl_xor(v, m, 32));
  return v;
}
__device__ __forceinline__ float rowsum16(float v) {
#pragma unroll
  for (int m = 8; m >= 1; m >>= 1) v += __shfl_xor(v, m, 32);
  return v;
}

// ---------------------------------------------------------------------------
// Kernel 1: QKV = x @ Wqkv^T  (M=4096, N=3072, K=1024), f32 -> f16 out,
// scattered into qkv[3][B][H][S][Dh].
// Block 256 threads (8 waves), tile 128(M) x 64(N), K chunk 32.
// ---------------------------------------------------------------------------
__global__ __launch_bounds__(256)
void qkv_gemm_kernel(const float* __restrict__ x,
                     const float* __restrict__ Wqkv,
                     _Float16* __restrict__ qkv) {
  __shared__ __align__(32) _Float16 At[128 * 32];  // [m][k]
  __shared__ __align__(32) _Float16 Bt[64 * 32];   // [n][k] (W is [n][k] already)
  const int tid  = threadIdx.x;
  const int wave = tid >> 5;
  const int lane = tid & 31;
  const int lm   = lane & 15;
  const int hi   = lane >> 4;
  const int m0   = blockIdx.y * 128;
  const int n0   = blockIdx.x * 64;

  v8f acc[4] = {};

  for (int k0 = 0; k0 < D_MODEL; k0 += 32) {
    // stage A: 128x32 f32 -> f16, 16 elems/thread
    {
      const int row = tid >> 1;
      const int col = (tid & 1) * 16;
      const float4* src = (const float4*)(x + (size_t)(m0 + row) * D_MODEL + k0 + col);
      _Float16* dst = &At[row * 32 + col];
#pragma unroll
      for (int u = 0; u < 4; ++u) {
        float4 f = src[u];
        dst[u * 4 + 0] = (_Float16)f.x; dst[u * 4 + 1] = (_Float16)f.y;
        dst[u * 4 + 2] = (_Float16)f.z; dst[u * 4 + 3] = (_Float16)f.w;
      }
      if (k0 + 32 < D_MODEL)
        __builtin_prefetch(x + (size_t)(m0 + row) * D_MODEL + k0 + 32 + col, 0, 1);
    }
    // stage B: 64x32 f32 -> f16, 8 elems/thread
    {
      const int row = tid >> 2;
      const int col = (tid & 3) * 8;
      const float4* src = (const float4*)(Wqkv + (size_t)(n0 + row) * D_MODEL + k0 + col);
      _Float16* dst = &Bt[row * 32 + col];
#pragma unroll
      for (int u = 0; u < 2; ++u) {
        float4 f = src[u];
        dst[u * 4 + 0] = (_Float16)f.x; dst[u * 4 + 1] = (_Float16)f.y;
        dst[u * 4 + 2] = (_Float16)f.z; dst[u * 4 + 3] = (_Float16)f.w;
      }
    }
    __syncthreads();

    v16h a = load_a_frag(&At[(wave * 16 + lm) * 32], hi);
#pragma unroll
    for (int j = 0; j < 4; ++j) {
      // B-fragment: lane holds col n=lm, K contiguous at hi*16
      v16h b = *(const v16h*)(&Bt[(j * 16 + lm) * 32 + hi * 16]);
      acc[j] = wmma16(a, b, acc[j]);
    }
    __syncthreads();
  }

  // scatter epilogue: (m,n) -> qkv[which][b][h][s][d], f16
#pragma unroll
  for (int j = 0; j < 4; ++j) {
#pragma unroll
    for (int r = 0; r < 8; ++r) {
      const int m = m0 + wave * 16 + r + hi * 8;
      const int n = n0 + j * 16 + lm;
      const int b = m >> 11, s = m & 2047;
      const int which = n >> 10, rem = n & 1023;
      const int h = rem >> 6, d = rem & 63;
      const size_t idx = (((size_t)which * BATCH * NHEAD + (size_t)b * NHEAD + h)
                          * S_LEN + s) * HDIM + d;
      qkv[idx] = (_Float16)acc[j][r];
    }
  }
}

// ---------------------------------------------------------------------------
// Kernel 2: causal flash attention per (b,h), 64 query rows per block.
// Block 128 threads (4 waves); each wave owns a 16-row q strip.
// K and V tiles staged [key][d] with vector b128 copies; PV B-fragments
// gathered with strided ds_load_u16 (cheaper than a scalar LDS transpose).
// ---------------------------------------------------------------------------
__global__ __launch_bounds__(128)
void attn_kernel(const _Float16* __restrict__ qkv,
                 _Float16* __restrict__ ctx) {
  __shared__ __align__(32) _Float16 Kt[32 * 64];      // [key][d]
  __shared__ __align__(32) _Float16 Vt[32 * 64];      // [key][d]
  __shared__ __align__(32) _Float16 Pt[4][16 * 32];   // per-wave P tile [m][key]
  const int tid  = threadIdx.x;
  const int wave = tid >> 5;
  const int lane = tid & 31;
  const int lm   = lane & 15;
  const int hi   = lane >> 4;
  const int bh   = blockIdx.y;            // b*NHEAD + h
  const int q0   = blockIdx.x * 64;

  const size_t headoff = (size_t)bh * S_LEN * HDIM;
  const _Float16* Q = qkv + headoff;
  const _Float16* K = qkv + (size_t)BATCH * NHEAD * S_LEN * HDIM + headoff;
  const _Float16* V = qkv + (size_t)2 * BATCH * NHEAD * S_LEN * HDIM + headoff;

  // Q strip -> two A fragments (Dh = 64 = 2 x K32), straight from global
  const _Float16* qrow = Q + (size_t)(q0 + wave * 16 + lm) * HDIM;
  const v16h aq0 = load_a_frag(qrow, hi);
  const v16h aq1 = load_a_frag(qrow + 32, hi);

  v8f acc[4] = {};
  float mrow[8], lrow[8];
#pragma unroll
  for (int r = 0; r < 8; ++r) { mrow[r] = -1e30f; lrow[r] = 0.0f; }

  const float scale = 0.125f;  // 1/sqrt(64)
  const int nkt = (q0 + 64) / 32;  // key tiles covering [0, q0+64)

  // each thread stages 16 contiguous halves of K and V per tile
  const int st_key = tid >> 2;          // 0..31
  const int st_d   = (tid & 3) * 16;    // 0,16,32,48

  for (int kt = 0; kt < nkt; ++kt) {
    const int k0 = kt * 32;
    __syncthreads();  // protect tiles from previous iteration
    // stage K and V tiles [32][64]: vector b128 copies only
    *(v16h*)(&Kt[st_key * 64 + st_d]) =
        *(const v16h*)(K + (size_t)(k0 + st_key) * HDIM + st_d);
    *(v16h*)(&Vt[st_key * 64 + st_d]) =
        *(const v16h*)(V + (size_t)(k0 + st_key) * HDIM + st_d);
    if (kt + 1 < nkt) {
      __builtin_prefetch(K + (size_t)(k0 + 32 + st_key) * HDIM + st_d, 0, 1);
      __builtin_prefetch(V + (size_t)(k0 + 32 + st_key) * HDIM + st_d, 0, 1);
    }
    __syncthreads();

    // scores: two 16x16 column groups, each = 2 WMMAs over Dh
    float p[2][8];
#pragma unroll
    for (int j = 0; j < 2; ++j) {
      v16h b0 = *(const v16h*)(&Kt[(j * 16 + lm) * 64 + hi * 16]);
      v16h b1 = *(const v16h*)(&Kt[(j * 16 + lm) * 64 + 32 + hi * 16]);
      v8f sc = {};
      sc = wmma16(aq0, b0, sc);
      sc = wmma16(aq1, b1, sc);
      const int key = k0 + j * 16 + lm;
#pragma unroll
      for (int r = 0; r < 8; ++r) {
        const int qg = q0 + wave * 16 + r + hi * 8;
        p[j][r] = (key > qg) ? -1e30f : sc[r] * scale;
      }
    }

    // online softmax per row (rows live across the 16-lane halves)
#pragma unroll
    for (int r = 0; r < 8; ++r) {
      float t = rowmax16(fmaxf(p[0][r], p[1][r]));
      const float mn = fmaxf(mrow[r], t);
      const float alpha = __expf(mrow[r] - mn);
      p[0][r] = __expf(p[0][r] - mn);
      p[1][r] = __expf(p[1][r] - mn);
      const float rs = rowsum16(p[0][r] + p[1][r]);
      lrow[r] = lrow[r] * alpha + rs;
      mrow[r] = mn;
#pragma unroll
      for (int j = 0; j < 4; ++j) acc[j][r] *= alpha;
      Pt[wave][(r + hi * 8) * 32 + lm]      = (_Float16)p[0][r];
      Pt[wave][(r + hi * 8) * 32 + 16 + lm] = (_Float16)p[1][r];
    }

    // P(16x32) @ V(32x64): per-wave LDS round-trip re-layouts P as A-fragment;
    // B-fragment gathered from Vt[key][d] with strided u16 loads.
    const v16h ap = load_a_frag(&Pt[wave][lm * 32], hi);
#pragma unroll
    for (int j = 0; j < 4; ++j) {
      const int d = j * 16 + lm;
      v16h bv;
#pragma unroll
      for (int i = 0; i < 16; ++i) bv[i] = Vt[(hi * 16 + i) * 64 + d];
      acc[j] = wmma16(ap, bv, acc[j]);
    }
  }

  // normalize + store ctx as f16 [B][S][H*Dh]
  const int b = bh >> 4, h = bh & 15;
#pragma unroll
  for (int r = 0; r < 8; ++r) {
    const float inv = 1.0f / lrow[r];
    const int qg = q0 + wave * 16 + r + hi * 8;
    const size_t rowbase = ((size_t)b * S_LEN + qg) * D_MODEL + h * HDIM;
#pragma unroll
    for (int j = 0; j < 4; ++j)
      ctx[rowbase + j * 16 + lm] = (_Float16)(acc[j][r] * inv);
  }
}

// ---------------------------------------------------------------------------
// Kernel 3: out = ctx @ Wout^T + bias  (M=4096, N=1024, K=1024), f32 out.
// ---------------------------------------------------------------------------
__global__ __launch_bounds__(256)
void out_gemm_kernel(const _Float16* __restrict__ ctx,
                     const float* __restrict__ Wout,
                     const float* __restrict__ bias,
                     float* __restrict__ out) {
  __shared__ __align__(32) _Float16 At[128 * 32];
  __shared__ __align__(32) _Float16 Bt[64 * 32];
  const int tid  = threadIdx.x;
  const int wave = tid >> 5;
  const int lane = tid & 31;
  const int lm   = lane & 15;
  const int hi   = lane >> 4;
  const int m0   = blockIdx.y * 128;
  const int n0   = blockIdx.x * 64;

  v8f acc[4] = {};

  for (int k0 = 0; k0 < D_MODEL; k0 += 32) {
    {  // stage A (already f16): 16 contiguous per thread
      const int row = tid >> 1;
      const int col = (tid & 1) * 16;
      *(v16h*)(&At[row * 32 + col]) =
          *(const v16h*)(ctx + (size_t)(m0 + row) * D_MODEL + k0 + col);
      if (k0 + 32 < D_MODEL)
        __builtin_prefetch(ctx + (size_t)(m0 + row) * D_MODEL + k0 + 32 + col, 0, 1);
    }
    {  // stage B: Wout[n][k] f32 -> f16
      const int row = tid >> 2;
      const int col = (tid & 3) * 8;
      const float4* src = (const float4*)(Wout + (size_t)(n0 + row) * D_MODEL + k0 + col);
      _Float16* dst = &Bt[row * 32 + col];
#pragma unroll
      for (int u = 0; u < 2; ++u) {
        float4 f = src[u];
        dst[u * 4 + 0] = (_Float16)f.x; dst[u * 4 + 1] = (_Float16)f.y;
        dst[u * 4 + 2] = (_Float16)f.z; dst[u * 4 + 3] = (_Float16)f.w;
      }
    }
    __syncthreads();

    v16h a = load_a_frag(&At[(wave * 16 + lm) * 32], hi);
#pragma unroll
    for (int j = 0; j < 4; ++j) {
      v16h b = *(const v16h*)(&Bt[(j * 16 + lm) * 32 + hi * 16]);
      acc[j] = wmma16(a, b, acc[j]);
    }
    __syncthreads();
  }

#pragma unroll
  for (int j = 0; j < 4; ++j) {
    const int n = n0 + j * 16 + lm;
    const float bb = bias[n];
#pragma unroll
    for (int r = 0; r < 8; ++r) {
      const int m = m0 + wave * 16 + r + hi * 8;
      out[(size_t)m * D_MODEL + n] = acc[j][r] + bb;
    }
  }
}

// ---------------------------------------------------------------------------
extern "C" void kernel_launch(void* const* d_in, const int* in_sizes, int n_in,
                              void* d_out, int out_size, void* d_ws, size_t ws_size,
                              hipStream_t stream) {
  const float* x    = (const float*)d_in[0];
  const float* Wqkv = (const float*)d_in[1];
  const float* Wout = (const float*)d_in[2];
  const float* bout = (const float*)d_in[3];
  float* out = (float*)d_out;

  _Float16* qkv = (_Float16*)d_ws;                 // 3*B*H*S*Dh f16 (~25.2 MB)
  _Float16* ctx = qkv + QKV_ELEMS;                 // B*S*D f16 (~8.4 MB)

  // 1) QKV projection: grid (N/64, M/128) = (48, 32)
  qkv_gemm_kernel<<<dim3(3 * D_MODEL / 64, BS_ROWS / 128), 256, 0, stream>>>(
      x, Wqkv, qkv);

  // 2) causal attention: grid (S/64, B*H) = (32, 32)
  attn_kernel<<<dim3(S_LEN / 64, BATCH * NHEAD), 128, 0, stream>>>(qkv, ctx);

  // 3) output projection + bias: grid (N/64, M/128) = (16, 32)
  out_gemm_kernel<<<dim3(D_MODEL / 64, BS_ROWS / 128), 256, 0, stream>>>(
      ctx, Wout, bout, out);
}
